// GraformerLayer_14285061226622
// MI455X (gfx1250) — matrix operand order repaired and checked by reading, not compiled, not run on hardware
//
#include <hip/hip_runtime.h>
#include <hip/hip_bf16.h>

// ===== CDNA5 (gfx1250) wave32 WMMA implementation of the Graformer layer =====
// Requires ws_size >= ~60 MB.

typedef __attribute__((ext_vector_type(16))) __bf16 v16bf;
typedef __attribute__((ext_vector_type(8)))  float  v8f;

static constexpr int Bn  = 8;
static constexpr int Nn  = 1024;
static constexpr int Dd  = 256;
static constexpr int Hh  = 8;
static constexpr int DH  = 32;
static constexpr int DFF = 1024;

// ---------------- bf16 helpers (RNE, bit-exact independent of host libs) ----
__device__ __forceinline__ __bf16 f2bf(float f) {
  unsigned u = __float_as_uint(f);
  unsigned r = (u + 0x7FFFu + ((u >> 16) & 1u)) >> 16;
  unsigned short h = (unsigned short)r;
  return __builtin_bit_cast(__bf16, h);
}
__device__ __forceinline__ float bf2f(__bf16 h) {
  unsigned short s = __builtin_bit_cast(unsigned short, h);
  return __uint_as_float(((unsigned)s) << 16);
}

// ---------------- WMMA fragment loaders (per ISA §7.12.2, 16-bit layouts) ----
// A: 16x32 tile, row-major with row stride lda (elements).
// lane = l: row = l&15; lanes 0-15 hold K 0..7 (v0..3) and 16..23 (v4..7),
// lanes 16-31 hold K 8..15 and 24..31; 2 packed K per VGPR.
__device__ __forceinline__ v16bf load_frag_A(const __bf16* A, int lda, int lane) {
  v16bf a;
  const int r = lane & 15, h = (lane >> 4) * 8;
  const __bf16* p = A + r * lda;
#pragma unroll
  for (int v = 0; v < 8; ++v) {
    const int kb = (v < 4 ? 2 * v : 8 + 2 * v) + h;   // 0,2,4,6 | 16,18,20,22 (+h)
    a[2 * v]     = p[kb];
    a[2 * v + 1] = p[kb + 1];
  }
  return a;
}
// B: 32x16 tile (KxN), element (k,n) at B[k*rs + n*cs].
// lanes 0-15: col n=lane, K 0..7,16..23; lanes 16-31: col n=lane-16, K 8..15,24..31.
__device__ __forceinline__ v16bf load_frag_B(const __bf16* B, int rs, int cs, int lane) {
  v16bf b;
  const int n = lane & 15, kh = (lane >> 4) * 16;
#pragma unroll
  for (int v = 0; v < 8; ++v) {
    const int k0 = (v < 4 ? 2 * v : 8 + 2 * v) + kh / 2;
    b[2 * v]     = B[k0 * rs + n * cs];
    b[2 * v + 1] = B[(k0 + 1) * rs + n * cs];
  }
  return b;
}

// ---------------- generic batched bf16 WMMA GEMM with fused epilogues -------
constexpr int EP_F32 = 0, EP_BF16 = 1, EP_BIAS_RELU_BF16 = 2,
              EP_BASE_BIAS_F32 = 3, EP_BASE_F32 = 4, EP_ACC_F32 = 5,
              EP_TX2_BF16 = 6;

struct GemmParams {
  const __bf16* A; long sA; int dA, mA, lda;      // A batch off = ((z/dA)%mA)*sA
  const __bf16* B; long sB; int dB, mB; int brs, bcs;
  float*  outF; __bf16* outB; long sOut; int ldo; // out batch off = z*sOut
  const float*  base; long sBase;                 // fp32 residual base
  const float*  bias;                             // per-column bias
  const __bf16* aux;  long sAux;                  // Tx0 for Chebyshev Tx2
  int M, N, K;
};

// Register-blocked: each wave computes a (16*MT) x (16*NT) tile.
// A fragments are reused NT times, B fragments MT times per K-step.
template <int EP, int MT, int NT>
__global__ void __launch_bounds__(32) gemm_wmma(GemmParams p) {
  const int lane = threadIdx.x & 31;
  const int n0 = blockIdx.x * 16 * NT;
  const int m0 = blockIdx.y * 16 * MT;
  const int z  = blockIdx.z;

  const __bf16* A  = p.A + (long)((z / p.dA) % p.mA) * p.sA + (long)m0 * p.lda;
  const __bf16* Bm = p.B + (long)((z / p.dB) % p.mB) * p.sB + (long)n0 * p.bcs;

  v8f acc[MT][NT];
#pragma unroll
  for (int mt = 0; mt < MT; ++mt)
#pragma unroll
    for (int nt = 0; nt < NT; ++nt) acc[mt][nt] = (v8f){};

  for (int k0 = 0; k0 < p.K; k0 += 32) {
    __builtin_prefetch(A + k0 + 64, 0, 0);        // global_prefetch_b8
    v16bf a[MT];
#pragma unroll
    for (int mt = 0; mt < MT; ++mt)
      a[mt] = load_frag_A(A + (long)(mt * 16) * p.lda + k0, p.lda, lane);
#pragma unroll
    for (int nt = 0; nt < NT; ++nt) {
      v16bf b = load_frag_B(Bm + (long)k0 * p.brs + nt * 16 * p.bcs,
                            p.brs, p.bcs, lane);
#pragma unroll
      for (int mt = 0; mt < MT; ++mt)
        acc[mt][nt] = __builtin_amdgcn_wmma_f32_16x16x32_bf16(
            false, a[mt], false, b, (short)0, acc[mt][nt], false, false);
    }
  }

  // C/D layout: lanes 0-15 -> rows 0..7 (v), lanes 16-31 -> rows 8..15; col=lane&15
  const int col = lane & 15, rb = (lane >> 4) * 8;
  const long ob = (long)z * p.sOut;
#pragma unroll
  for (int mt = 0; mt < MT; ++mt) {
#pragma unroll
    for (int nt = 0; nt < NT; ++nt) {
#pragma unroll
      for (int v = 0; v < 8; ++v) {
        const int r = m0 + mt * 16 + rb + v, c = n0 + nt * 16 + col;
        const long oi = ob + (long)r * p.ldo + c;
        float val = acc[mt][nt][v];
        if constexpr (EP == EP_F32) {
          p.outF[oi] = val;
        } else if constexpr (EP == EP_BF16) {
          p.outB[oi] = f2bf(val);
        } else if constexpr (EP == EP_BIAS_RELU_BF16) {
          val += p.bias[c];
          p.outB[oi] = f2bf(val > 0.f ? val : 0.f);
        } else if constexpr (EP == EP_BASE_BIAS_F32) {
          p.outF[oi] = p.base[(long)z * p.sBase + (long)r * p.ldo + c] + val + p.bias[c];
        } else if constexpr (EP == EP_BASE_F32) {
          p.outF[oi] = p.base[(long)z * p.sBase + (long)r * p.ldo + c] + val;
        } else if constexpr (EP == EP_ACC_F32) {
          p.outF[oi] += val;
        } else { // EP_TX2_BF16: Tx2 = 2*L@Tx1 - Tx0
          p.outB[oi] =
              f2bf(2.f * val - bf2f(p.aux[(long)z * p.sAux + (long)r * p.ldo + c]));
        }
      }
    }
  }
}

// ---------------- fused attention: scores -> softmax -> attn@G + residual ---
// One wave per (b, h, 16-row strip). dh==32 == one WMMA K step.
// Scores strip (16x1024 f32 = 64 KB) lives in LDS; probabilities are written
// back in place as bf16 (write byte offsets never pass unread score bytes;
// single wave => LDS ops are in order).
__global__ void __launch_bounds__(32) attn_fused(const __bf16* __restrict__ G,
                                                 const float*  __restrict__ Z,
                                                 float*        __restrict__ Z1) {
  __shared__ float sc[16 * Nn];                       // 64 KB
  __bf16* pr = (__bf16*)sc;                           // aliased probabilities

  const int lane = threadIdx.x & 31;
  const int m0 = blockIdx.x * 16;
  const int h  = blockIdx.y;
  const int b  = blockIdx.z;
  const __bf16* Gq = G + ((long)(b * Hh + h)) * Nn * DH;
  const float scale = 0.17677669529663687f;           // 1/sqrt(dh)

  // scores strip = (G_rows x G^T) * scale
  const v16bf aQ = load_frag_A(Gq + (long)m0 * DH, DH, lane);
  const int col = lane & 15, rb = (lane >> 4) * 8;
  for (int nt = 0; nt < Nn / 16; ++nt) {
    const int n0 = nt * 16;
    v16bf bK = load_frag_B(Gq + (long)n0 * DH, /*rs=*/1, /*cs=*/DH, lane); // G^T
    v8f c = {};
    c = __builtin_amdgcn_wmma_f32_16x16x32_bf16(false, aQ, false, bK,
                                                (short)0, c, false, false);
#pragma unroll
    for (int v = 0; v < 8; ++v) sc[(rb + v) * Nn + n0 + col] = c[v] * scale;
  }
  asm volatile("s_wait_dscnt 0" ::: "memory");

  // wave-cooperative softmax per row, emit bf16 probabilities in place
  for (int r = 0; r < 16; ++r) {
    float* row = sc + r * Nn;
    float mx = -3.0e38f;
    for (int i = lane; i < Nn; i += 32) mx = fmaxf(mx, row[i]);
#pragma unroll
    for (int o = 16; o; o >>= 1) mx = fmaxf(mx, __shfl_xor(mx, o, 32));
    float sum = 0.f;
    for (int i = lane; i < Nn; i += 32) {
      float e = __expf(row[i] - mx);
      row[i] = e;
      sum += e;
    }
#pragma unroll
    for (int o = 16; o; o >>= 1) sum += __shfl_xor(sum, o, 32);
    const float inv = 1.f / sum;
    for (int i = lane; i < Nn; i += 32) pr[r * Nn + i] = f2bf(row[i] * inv);
  }
  asm volatile("s_wait_dscnt 0" ::: "memory");

  // out(16x32) = attn(16x1024) @ G(1024x32); write Z1 = Z + out (head slice)
  v8f o0 = {}, o1 = {};
  for (int k0 = 0; k0 < Nn; k0 += 32) {
    v16bf aP = load_frag_A((const __bf16*)pr + k0, Nn, lane);
    v16bf b0 = load_frag_B(Gq + (long)k0 * DH + 0,  DH, 1, lane);
    v16bf b1 = load_frag_B(Gq + (long)k0 * DH + 16, DH, 1, lane);
    o0 = __builtin_amdgcn_wmma_f32_16x16x32_bf16(false, aP, false, b0,
                                                 (short)0, o0, false, false);
    o1 = __builtin_amdgcn_wmma_f32_16x16x32_bf16(false, aP, false, b1,
                                                 (short)0, o1, false, false);
  }
  const float* Zp = Z  + ((long)b * Nn + m0) * Dd + h * DH;
  float*       Op = Z1 + ((long)b * Nn + m0) * Dd + h * DH;
#pragma unroll
  for (int v = 0; v < 8; ++v) {
    const int r = rb + v;
    Op[r * Dd + col]      = Zp[r * Dd + col]      + o0[v];
    Op[r * Dd + 16 + col] = Zp[r * Dd + 16 + col] + o1[v];
  }
}

// ---------------- LayerNorm over d=256, one wave per row, bf16 out ----------
__global__ void __launch_bounds__(256) layernorm_bf16(const float* __restrict__ X,
                                                      const float* __restrict__ g,
                                                      const float* __restrict__ bta,
                                                      __bf16* __restrict__ Y, int rows) {
  const int lane = threadIdx.x & 31;
  const int row = blockIdx.x * 8 + (threadIdx.x >> 5);
  if (row >= rows) return;
  const float4* x4 = (const float4*)(X + (long)row * Dd);
  const float4 a = x4[lane], c = x4[lane + 32];
  float s = a.x + a.y + a.z + a.w + c.x + c.y + c.z + c.w;
  float q = a.x * a.x + a.y * a.y + a.z * a.z + a.w * a.w +
            c.x * c.x + c.y * c.y + c.z * c.z + c.w * c.w;
#pragma unroll
  for (int o = 16; o; o >>= 1) { s += __shfl_xor(s, o, 32); q += __shfl_xor(q, o, 32); }
  const float mu = s * (1.f / Dd);
  const float rstd = rsqrtf(q * (1.f / Dd) - mu * mu + 1e-5f);
  const float4* g4 = (const float4*)g;
  const float4* b4 = (const float4*)bta;
  const float4 g0 = g4[lane], g1 = g4[lane + 32], b0 = b4[lane], b1 = b4[lane + 32];
  __bf16* y = Y + (long)row * Dd;
  const int i0 = lane * 4, i1 = (lane + 32) * 4;
  y[i0 + 0] = f2bf((a.x - mu) * rstd * g0.x + b0.x);
  y[i0 + 1] = f2bf((a.y - mu) * rstd * g0.y + b0.y);
  y[i0 + 2] = f2bf((a.z - mu) * rstd * g0.z + b0.z);
  y[i0 + 3] = f2bf((a.w - mu) * rstd * g0.w + b0.w);
  y[i1 + 0] = f2bf((c.x - mu) * rstd * g1.x + b1.x);
  y[i1 + 1] = f2bf((c.y - mu) * rstd * g1.y + b1.y);
  y[i1 + 2] = f2bf((c.z - mu) * rstd * g1.z + b1.z);
  y[i1 + 3] = f2bf((c.w - mu) * rstd * g1.w + b1.w);
}

// ---------------- degree^-1/2 of A_hat = adj + I, one wave per row ----------
__global__ void __launch_bounds__(256) degree_rsqrt(const float* __restrict__ adj,
                                                    float* __restrict__ dinv, int rows) {
  const int lane = threadIdx.x & 31;
  const int row = blockIdx.x * 8 + (threadIdx.x >> 5);
  if (row >= rows) return;
  const float* a = adj + (long)row * Nn;
  float s = (lane == 0) ? 1.0f : 0.0f;                // self loop
  for (int i = lane; i < Nn; i += 32) s += a[i];
#pragma unroll
  for (int o = 16; o; o >>= 1) s += __shfl_xor(s, o, 32);
  if (lane == 0) dinv[row] = rsqrtf(s);
}

// ---------------- A_norm = D^-1/2 (adj + I) D^-1/2 -> bf16 ------------------
__global__ void build_anorm(const float* __restrict__ adj,
                            const float* __restrict__ dinv,
                            __bf16* __restrict__ an) {
  const long total = (long)Bn * Nn * Nn;
  for (long i = blockIdx.x * (long)blockDim.x + threadIdx.x; i < total;
       i += (long)gridDim.x * blockDim.x) {
    const int b = (int)(i >> 20);
    const int rem = (int)(i & ((1 << 20) - 1));
    const int n = rem >> 10, m = rem & (Nn - 1);
    const float v = adj[i] + (n == m ? 1.f : 0.f);
    an[i] = f2bf(v * dinv[b * Nn + n] * dinv[b * Nn + m]);
  }
}

// ---------------- fp32 -> bf16 cast ----------------------------------------
__global__ void f32_to_bf16(const float* __restrict__ x, __bf16* __restrict__ y, long n) {
  for (long i = blockIdx.x * (long)blockDim.x + threadIdx.x; i < n;
       i += (long)gridDim.x * blockDim.x)
    y[i] = f2bf(x[i]);
}

// ============================ host orchestration ============================
extern "C" void kernel_launch(void* const* d_in, const int* in_sizes, int n_in,
                              void* d_out, int out_size, void* d_ws, size_t ws_size,
                              hipStream_t stream) {
  (void)in_sizes; (void)n_in; (void)out_size; (void)ws_size;
  const float* Z    = (const float*)d_in[0];
  const float* adj  = (const float*)d_in[1];
  const float* lap  = (const float*)d_in[2];
  const float* ln1g = (const float*)d_in[3];
  const float* ln1b = (const float*)d_in[4];
  const float* ln2g = (const float*)d_in[5];
  const float* ln2b = (const float*)d_in[6];
  const float* ln3g = (const float*)d_in[7];
  const float* ln3b = (const float*)d_in[8];
  const float* Wh   = (const float*)d_in[9];
  const float* W1   = (const float*)d_in[10];
  const float* b1   = (const float*)d_in[11];
  const float* W2   = (const float*)d_in[12];
  const float* b2   = (const float*)d_in[13];
  const float* Th   = (const float*)d_in[14];

  char* ws = (char*)d_ws;
  size_t off = 0;
  auto alloc = [&](size_t bytes) {
    char* p = ws + off;
    off = (off + bytes + 255) & ~(size_t)255;
    return p;
  };
  // 16 MB region reused: A_norm -> FFN hidden -> bf16 laplacian (sequential lives)
  __bf16* shared16 = (__bf16*)alloc((size_t)Bn * Nn * Nn * 2);
  __bf16* zn1  = (__bf16*)alloc((size_t)Bn * Nn * Dd * 2);   // reused as zn2
  float*  dinv = (float*) alloc((size_t)Bn * Nn * 4);
  __bf16* whb  = (__bf16*)alloc((size_t)Hh * Dd * DH * 2);
  __bf16* xb   = (__bf16*)alloc((size_t)Bn * Hh * Nn * DH * 2);
  __bf16* gb   = (__bf16*)alloc((size_t)Bn * Hh * Nn * DH * 2);
  float*  z1   = (float*) alloc((size_t)Bn * Nn * Dd * 4);
  __bf16* w1b  = (__bf16*)alloc((size_t)Dd * DFF * 2);
  __bf16* w2b  = (__bf16*)alloc((size_t)DFF * Dd * 2);
  float*  z2   = (float*) alloc((size_t)Bn * Nn * Dd * 4);
  __bf16* tx0  = (__bf16*)alloc((size_t)Bn * Nn * Dd * 2);
  __bf16* tx1  = (__bf16*)alloc((size_t)Bn * Nn * Dd * 2);
  __bf16* tx2  = (__bf16*)alloc((size_t)Bn * Nn * Dd * 2);
  __bf16* thb  = (__bf16*)alloc((size_t)3 * Dd * Dd * 2);

  // ---- stage 1: LN1, degrees, normalized adjacency, head weights ----------
  layernorm_bf16<<<1024, 256, 0, stream>>>(Z, ln1g, ln1b, zn1, Bn * Nn);
  degree_rsqrt<<<1024, 256, 0, stream>>>(adj, dinv, Bn * Nn);
  build_anorm<<<4096, 256, 0, stream>>>(adj, dinv, shared16);
  f32_to_bf16<<<256, 256, 0, stream>>>(Wh, whb, (long)Hh * Dd * DH);

  // X[b,h] = Zn1[b] @ W_heads[h]   (M=1024, N=32, K=256)  32x32 per wave
  GemmParams px{};
  px.A = zn1; px.sA = (long)Nn * Dd; px.dA = Hh; px.mA = Bn; px.lda = Dd;
  px.B = whb; px.sB = (long)Dd * DH; px.dB = 1;  px.mB = Hh; px.brs = DH; px.bcs = 1;
  px.outB = xb; px.sOut = (long)Nn * DH; px.ldo = DH;
  px.M = Nn; px.N = DH; px.K = Dd;
  gemm_wmma<EP_BF16, 2, 2><<<dim3(1, Nn / 32, Bn * Hh), 32, 0, stream>>>(px);

  // G[b,h] = A_norm[b] @ X[b,h]    (M=1024, N=32, K=1024) 32x32 per wave
  GemmParams pg{};
  pg.A = shared16; pg.sA = (long)Nn * Nn; pg.dA = Hh; pg.mA = Bn; pg.lda = Nn;
  pg.B = xb; pg.sB = (long)Nn * DH; pg.dB = 1; pg.mB = 1 << 20; pg.brs = DH; pg.bcs = 1;
  pg.outB = gb; pg.sOut = (long)Nn * DH; pg.ldo = DH;
  pg.M = Nn; pg.N = DH; pg.K = Nn;
  gemm_wmma<EP_BF16, 2, 2><<<dim3(1, Nn / 32, Bn * Hh), 32, 0, stream>>>(pg);

  // fused scores/softmax/attn@G + residual -> z1
  attn_fused<<<dim3(Nn / 16, Hh, Bn), 32, 0, stream>>>(gb, Z, z1);

  // ---- stage 2: FFN ---------------------------------------------------------
  __bf16* zn2 = zn1;
  layernorm_bf16<<<1024, 256, 0, stream>>>(z1, ln2g, ln2b, zn2, Bn * Nn);
  f32_to_bf16<<<256, 256, 0, stream>>>(W1, w1b, (long)Dd * DFF);
  __bf16* hid = shared16;

  GemmParams pf1{};
  pf1.A = zn2; pf1.sA = 0; pf1.dA = 1; pf1.mA = 1; pf1.lda = Dd;
  pf1.B = w1b; pf1.sB = 0; pf1.dB = 1; pf1.mB = 1; pf1.brs = DFF; pf1.bcs = 1;
  pf1.outB = hid; pf1.sOut = 0; pf1.ldo = DFF; pf1.bias = b1;
  pf1.M = Bn * Nn; pf1.N = DFF; pf1.K = Dd;
  gemm_wmma<EP_BIAS_RELU_BF16, 2, 4>
      <<<dim3(DFF / 64, Bn * Nn / 32, 1), 32, 0, stream>>>(pf1);

  f32_to_bf16<<<256, 256, 0, stream>>>(W2, w2b, (long)DFF * Dd);
  GemmParams pf2{};
  pf2.A = hid; pf2.sA = 0; pf2.dA = 1; pf2.mA = 1; pf2.lda = DFF;
  pf2.B = w2b; pf2.sB = 0; pf2.dB = 1; pf2.mB = 1; pf2.brs = Dd; pf2.bcs = 1;
  pf2.outF = z2; pf2.sOut = 0; pf2.ldo = Dd;
  pf2.base = z1; pf2.sBase = 0; pf2.bias = b2;
  pf2.M = Bn * Nn; pf2.N = Dd; pf2.K = DFF;
  gemm_wmma<EP_BASE_BIAS_F32, 2, 4>
      <<<dim3(Dd / 64, Bn * Nn / 32, 1), 32, 0, stream>>>(pf2);

  // ---- stage 3: Chebyshev graph conv ---------------------------------------
  layernorm_bf16<<<1024, 256, 0, stream>>>(z2, ln3g, ln3b, tx0, Bn * Nn);
  __bf16* lapb = shared16;
  f32_to_bf16<<<8192, 256, 0, stream>>>(lap, lapb, (long)Bn * Nn * Nn);
  f32_to_bf16<<<256, 256, 0, stream>>>(Th, thb, (long)3 * Dd * Dd);

  // Tx1 = L @ Tx0              (per b: M=1024, N=256, K=1024)
  GemmParams pt1{};
  pt1.A = lapb; pt1.sA = (long)Nn * Nn; pt1.dA = 1; pt1.mA = Bn; pt1.lda = Nn;
  pt1.B = tx0; pt1.sB = (long)Nn * Dd; pt1.dB = 1; pt1.mB = Bn; pt1.brs = Dd; pt1.bcs = 1;
  pt1.outB = tx1; pt1.sOut = (long)Nn * Dd; pt1.ldo = Dd;
  pt1.M = Nn; pt1.N = Dd; pt1.K = Nn;
  gemm_wmma<EP_BF16, 2, 4><<<dim3(Dd / 64, Nn / 32, Bn), 32, 0, stream>>>(pt1);

  // Tx2 = 2 * L @ Tx1 - Tx0
  GemmParams pt2 = pt1;
  pt2.B = tx1; pt2.outB = tx2; pt2.aux = tx0; pt2.sAux = (long)Nn * Dd;
  gemm_wmma<EP_TX2_BF16, 2, 4><<<dim3(Dd / 64, Nn / 32, Bn), 32, 0, stream>>>(pt2);

  // d_out = z2 + Tx0@Th0; then += Tx1@Th1; then += Tx2@Th2
  float* out = (float*)d_out;
  GemmParams pc0{};
  pc0.A = tx0; pc0.sA = (long)Nn * Dd; pc0.dA = 1; pc0.mA = Bn; pc0.lda = Dd;
  pc0.B = thb; pc0.sB = 0; pc0.dB = 1; pc0.mB = 1; pc0.brs = Dd; pc0.bcs = 1;
  pc0.outF = out; pc0.sOut = (long)Nn * Dd; pc0.ldo = Dd;
  pc0.base = z2; pc0.sBase = (long)Nn * Dd;
  pc0.M = Nn; pc0.N = Dd; pc0.K = Dd;
  gemm_wmma<EP_BASE_F32, 2, 4><<<dim3(Dd / 64, Nn / 32, Bn), 32, 0, stream>>>(pc0);

  GemmParams pc1 = pc0; pc1.A = tx1; pc1.B = thb + Dd * Dd;
  gemm_wmma<EP_ACC_F32, 2, 4><<<dim3(Dd / 64, Nn / 32, Bn), 32, 0, stream>>>(pc1);
  GemmParams pc2 = pc0; pc2.A = tx2; pc2.B = thb + 2 * Dd * Dd;
  gemm_wmma<EP_ACC_F32, 2, 4><<<dim3(Dd / 64, Nn / 32, Bn), 32, 0, stream>>>(pc2);
}
